// DeltaNetLayer_25108378812593
// MI455X (gfx1250) — matrix operand order repaired and checked by reading, not compiled, not run on hardware
//
#include <hip/hip_runtime.h>

// ---------------------------------------------------------------------------
// DeltaNet layer for MI455X (gfx1250, wave32, WMMA).
//
// Math: in the reference scan, all rows of h stay identical (h0=0 and the
// update adds the same row to every row), so h[j,k] == H[k] with
//   H = beta_t*H + vsum_t*k_t,  o_t = (sum_j q_t[j]) * H.
// Hence only head-sums of q and v are needed -> q/v projections collapse to
// 16384x1024x16 GEMMs.  Full bf16 WMMA GEMMs remain for k = x@Wk, out = o@Wo.
// ---------------------------------------------------------------------------

typedef __attribute__((ext_vector_type(16))) __bf16 v16bf;
typedef __attribute__((ext_vector_type(8)))  float  v8f;
typedef int v4i_t __attribute__((vector_size(16)));

#define BQ 4
#define SQ 4096
#define HID 1024
#define NH 16
#define DH 64
#define MROWS (BQ * SQ)   // 16384

// CDNA5 async global->LDS path (ASYNCcnt), guarded so either toolchain builds.
#if defined(__gfx1250__) && \
    __has_builtin(__builtin_amdgcn_global_load_async_to_lds_b128) && \
    __has_builtin(__builtin_amdgcn_s_wait_asynccnt)
#define USE_ASYNC 1
#else
#define USE_ASYNC 0
#endif

#define AS1 __attribute__((address_space(1)))
#define AS3 __attribute__((address_space(3)))

__device__ inline unsigned short f2bf(float f) {
  unsigned u = __float_as_uint(f);
  u += 0x7FFFu + ((u >> 16) & 1u);   // round-to-nearest-even
  return (unsigned short)(u >> 16);
}
__device__ inline float bf2f(unsigned short h) {
  return __uint_as_float(((unsigned)h) << 16);
}

// 16-byte global->LDS copy: async DMA on CDNA5, plain copy otherwise.
__device__ inline void copy16(const unsigned short* g, unsigned short* l) {
#if USE_ASYNC
  __builtin_amdgcn_global_load_async_to_lds_b128(
      (AS1 v4i_t*)g, (AS3 v4i_t*)l, 0, 0);
#else
  *(uint4*)l = *(const uint4*)g;
#endif
}

// ---------------- cast x (f32 -> bf16), 4 elements/thread -------------------
__global__ void cast_x_kernel(const float* __restrict__ x,
                              unsigned short* __restrict__ xb, int n4) {
  int i = blockIdx.x * blockDim.x + threadIdx.x;
  if (i >= n4) return;
  float4 v = ((const float4*)x)[i];
  ushort4 o;
  o.x = f2bf(v.x); o.y = f2bf(v.y); o.z = f2bf(v.z); o.w = f2bf(v.w);
  ((ushort4*)xb)[i] = o;
}

// ------------- transpose+cast weight: WT[n*K+k] = bf16(W[k*N+n]) ------------
__global__ void wT_kernel(const float* __restrict__ W,
                          unsigned short* __restrict__ WT, int N, int K) {
  int i = blockIdx.x * blockDim.x + threadIdx.x;
  if (i >= N * K) return;
  int n = i / K, k = i - n * K;
  WT[i] = f2bf(W[(size_t)k * N + n]);
}

// ---- per-head column sum: Ws[j*16+h] = sum_{d<64} W[j*1024 + h*64 + d] -----
__global__ void headsum_kernel(const float* __restrict__ W,
                               float* __restrict__ Ws) {
  int g = blockIdx.x * blockDim.x + threadIdx.x;
  if (g >= HID * NH) return;
  int j = g >> 4, h = g & 15;
  const float* p = W + (size_t)j * HID + h * DH;
  float s = 0.f;
#pragma unroll 8
  for (int d = 0; d < DH; ++d) s += p[d];
  Ws[g] = s;
}

// ------- small fused projection: qsum | vsum | beta  (f32, N=48) ------------
__global__ void smallproj_kernel(const float* __restrict__ x,
                                 const float* __restrict__ Wqs,
                                 const float* __restrict__ Wvs,
                                 const float* __restrict__ Wbeta,
                                 const float* __restrict__ b_beta,
                                 float* __restrict__ qs,
                                 float* __restrict__ vs,
                                 float* __restrict__ beta) {
  __shared__ float xs[4][HID];
  int row0 = blockIdx.x * 4;
  int ty = threadIdx.y, tx = threadIdx.x;   // blockDim (64,4)
  for (int k = tx; k < HID; k += 64)
    xs[ty][k] = x[(size_t)(row0 + ty) * HID + k];
  __syncthreads();
  if (tx >= 48) return;
  int c = tx;
  float acc = 0.f;
  if (c < 16) {
    for (int k = 0; k < HID; ++k) acc += xs[ty][k] * Wqs[k * 16 + c];
    qs[(size_t)(row0 + ty) * 16 + c] = acc;
  } else if (c < 32) {
    int h = c - 16;
    for (int k = 0; k < HID; ++k) acc += xs[ty][k] * Wvs[k * 16 + h];
    vs[(size_t)(row0 + ty) * 16 + h] = acc;
  } else {
    int h = c - 32;
    for (int k = 0; k < HID; ++k) acc += xs[ty][k] * Wbeta[k * 16 + h];
    beta[(size_t)(row0 + ty) * 16 + h] = 1.f / (1.f + __expf(-(acc + b_beta[h])));
  }
}

// --------------------------- WMMA bf16 GEMM ---------------------------------
// C(MxN,f32 accum) = A(MxK,bf16 row-major) * BT(NxK,bf16 row-major)^T
// block tile 128x128, BK=32, 8 waves (2x4), wave tile 64x32 (4x2 wmma tiles).
// LDS double-buffered; next tile staged via async-to-LDS while WMMAs run.
#define BM 128
#define BN 128
#define BK 32
#define LDST 48   // LDS row stride in ushorts (96B, keeps 16B alignment)

struct Frag32B { uint4 a, b; };

template <bool BF16OUT>
__global__ __launch_bounds__(256) void gemm_bf16_kernel(
    const unsigned short* __restrict__ A,
    const unsigned short* __restrict__ BT,
    unsigned short* __restrict__ Cb,      // used when BF16OUT
    float* __restrict__ Cf,               // used when !BF16OUT
    const float* __restrict__ bias,       // used when !BF16OUT (len N)
    const float* __restrict__ resid,      // used when !BF16OUT (MxN)
    int M, int N, int K) {
  __shared__ __align__(16) unsigned short As[2][BM * LDST];
  __shared__ __align__(16) unsigned short Bs[2][BN * LDST];

  const int tid  = threadIdx.x;
  const int lane = tid & 31;
  const int wave = tid >> 5;
  const int wm = wave >> 2;        // 0..1 -> 64 rows each
  const int wn = wave & 3;         // 0..3 -> 32 cols each
  const int rowBlk = blockIdx.y * BM;
  const int colBlk = blockIdx.x * BN;
  const int hl  = lane >> 4;       // half-wave select
  const int l15 = lane & 15;

  // per-thread staging coordinates: 512 16B chunks per tile, 2 per thread
  int sr[2], sc[2];
#pragma unroll
  for (int i = 0; i < 2; ++i) {
    int chunk = tid + i * 256;
    sr[i] = chunk >> 2;
    sc[i] = (chunk & 3) << 3;
  }

  v8f acc[4][2];
#pragma unroll
  for (int mi = 0; mi < 4; ++mi)
#pragma unroll
    for (int ni = 0; ni < 2; ++ni)
#pragma unroll
      for (int e = 0; e < 8; ++e) acc[mi][ni][e] = 0.f;

  // prologue: stage k0 = 0 into buffer 0
#pragma unroll
  for (int i = 0; i < 2; ++i) {
    copy16(&A[(size_t)(rowBlk + sr[i]) * K + sc[i]], &As[0][sr[i] * LDST + sc[i]]);
    copy16(&BT[(size_t)(colBlk + sr[i]) * K + sc[i]], &Bs[0][sr[i] * LDST + sc[i]]);
  }
#if USE_ASYNC
  __builtin_amdgcn_s_wait_asynccnt(0);
#endif
  __syncthreads();

  int buf = 0;
  for (int k0 = 0; k0 < K; k0 += BK) {
    const int kn = k0 + BK;
    if (kn < K) {
      // stage next tile into the other buffer (async DMA overlaps WMMA)
#pragma unroll
      for (int i = 0; i < 2; ++i) {
        copy16(&A[(size_t)(rowBlk + sr[i]) * K + kn + sc[i]],
               &As[buf ^ 1][sr[i] * LDST + sc[i]]);
        copy16(&BT[(size_t)(colBlk + sr[i]) * K + kn + sc[i]],
               &Bs[buf ^ 1][sr[i] * LDST + sc[i]]);
      }
      if (kn + BK < K) {
        // warm GL2 two tiles ahead
#pragma unroll
        for (int i = 0; i < 2; ++i) {
          __builtin_prefetch(&A[(size_t)(rowBlk + sr[i]) * K + kn + BK + sc[i]], 0, 1);
          __builtin_prefetch(&BT[(size_t)(colBlk + sr[i]) * K + kn + BK + sc[i]], 0, 1);
        }
      }
    }

    // A fragments: 16-bit A 16x32 layout -> lane l15 is M, K halves per hl
    v16bf af[4];
#pragma unroll
    for (int mi = 0; mi < 4; ++mi) {
      const unsigned short* p =
          &As[buf][(wm * 64 + mi * 16 + l15) * LDST + hl * 8];
      Frag32B f;
      f.a = *(const uint4*)p;          // K = base .. base+7
      f.b = *(const uint4*)(p + 16);   // K = base+16 .. base+23
      af[mi] = __builtin_bit_cast(v16bf, f);
    }
    // B fragments: 32x16 B layout -> lane l15 is N, K halves per hl (same as A)
    v16bf bfr[2];
#pragma unroll
    for (int ni = 0; ni < 2; ++ni) {
      const unsigned short* p =
          &Bs[buf][(wn * 32 + ni * 16 + l15) * LDST + hl * 8];
      Frag32B f;
      f.a = *(const uint4*)p;          // K = base .. base+7
      f.b = *(const uint4*)(p + 16);   // K = base+16 .. base+23
      bfr[ni] = __builtin_bit_cast(v16bf, f);
    }
#pragma unroll
    for (int mi = 0; mi < 4; ++mi)
#pragma unroll
      for (int ni = 0; ni < 2; ++ni)
        acc[mi][ni] = __builtin_amdgcn_wmma_f32_16x16x32_bf16(
            false, af[mi], false, bfr[ni], (short)0, acc[mi][ni], false, false);

#if USE_ASYNC
    if (kn < K) __builtin_amdgcn_s_wait_asynccnt(0);
#endif
    __syncthreads();
    buf ^= 1;
  }

  // Epilogue (compile-time selected). C/D layout: VGPR r -> M = r + 8*hl, N = l15.
#pragma unroll
  for (int mi = 0; mi < 4; ++mi)
#pragma unroll
    for (int ni = 0; ni < 2; ++ni) {
      const int colg = colBlk + wn * 32 + ni * 16 + l15;
      if (BF16OUT) {
#pragma unroll
        for (int r = 0; r < 8; ++r) {
          int rowg = rowBlk + wm * 64 + mi * 16 + r + hl * 8;
          Cb[(size_t)rowg * N + colg] = f2bf(acc[mi][ni][r]);
        }
      } else {
        const float bv = bias[colg];
#pragma unroll
        for (int r = 0; r < 8; ++r) {
          int rowg = rowBlk + wm * 64 + mi * 16 + r + hl * 8;
          float v = acc[mi][ni][r] + bv + resid[(size_t)rowg * N + colg];
          Cf[(size_t)rowg * N + colg] = v;
        }
      }
    }
}

// -------------------- 3-phase chunked linear scan ---------------------------
// Recurrence per (b,h,d): H = beta*H + vsum*k ; o = qsum*H.  Linear, so split
// S=4096 into 64 chunks of 64: chunk-local scan / chunk combine / replay.
#define NC 64
#define LC 64

__global__ void scan_p1(const unsigned short* __restrict__ kb,
                        const float* __restrict__ vs,
                        const float* __restrict__ beta,
                        float* __restrict__ Lend,
                        float* __restrict__ Aprod) {
  int bh = blockIdx.x >> 6;        // (b*16+h)
  int c  = blockIdx.x & 63;
  int b = bh >> 4, h = bh & 15;
  int d = threadIdx.x;             // 64
  float H = 0.f, A = 1.f;
  size_t base = ((size_t)b * SQ + (size_t)c * LC) * NH + h;
  for (int t = 0; t < LC; ++t) {
    size_t idx = base + (size_t)t * NH;
    float bt = beta[idx];
    float kv = bf2f(kb[idx * DH + d]);
    H = fmaf(bt, H, vs[idx] * kv);
    A *= bt;
  }
  Lend[(size_t)blockIdx.x * DH + d] = H;
  if (d == 0) Aprod[blockIdx.x] = A;
}

__global__ void scan_p2(const float* __restrict__ Lend,
                        const float* __restrict__ Aprod,
                        float* __restrict__ Hcs) {
  int bh = blockIdx.x;             // 64 blocks
  int d = threadIdx.x;             // 64
  float H = 0.f;
  for (int c = 0; c < NC; ++c) {
    size_t i = ((size_t)bh * NC + c) * DH + d;
    Hcs[i] = H;
    H = Aprod[bh * NC + c] * H + Lend[i];
  }
}

__global__ void scan_p3(const unsigned short* __restrict__ kb,
                        const float* __restrict__ qs,
                        const float* __restrict__ vs,
                        const float* __restrict__ beta,
                        const float* __restrict__ Hcs,
                        unsigned short* __restrict__ ob) {
  int bh = blockIdx.x >> 6;
  int c  = blockIdx.x & 63;
  int b = bh >> 4, h = bh & 15;
  int d = threadIdx.x;
  float H = Hcs[(size_t)blockIdx.x * DH + d];
  size_t base = ((size_t)b * SQ + (size_t)c * LC) * NH + h;
  for (int t = 0; t < LC; ++t) {
    size_t idx = base + (size_t)t * NH;
    float bt = beta[idx];
    float kv = bf2f(kb[idx * DH + d]);
    H = fmaf(bt, H, vs[idx] * kv);
    ob[idx * DH + d] = f2bf(qs[idx] * H);
  }
}

// ------------------------- in-place LayerNorm -------------------------------
__global__ void ln_kernel(float* __restrict__ y,
                          const float* __restrict__ g,
                          const float* __restrict__ bta) {
  __shared__ float s1[256], s2[256];
  int tid = threadIdx.x;
  float* yr = y + (size_t)blockIdx.x * HID;
  float v[4]; float s = 0.f, sq = 0.f;
#pragma unroll
  for (int j = 0; j < 4; ++j) {
    v[j] = yr[tid + j * 256];
    s += v[j]; sq += v[j] * v[j];
  }
  s1[tid] = s; s2[tid] = sq; __syncthreads();
  for (int o = 128; o > 0; o >>= 1) {
    if (tid < o) { s1[tid] += s1[tid + o]; s2[tid] += s2[tid + o]; }
    __syncthreads();
  }
  float mu  = s1[0] * (1.f / HID);
  float var = s2[0] * (1.f / HID) - mu * mu;
  float rs  = rsqrtf(var + 1e-5f);
#pragma unroll
  for (int j = 0; j < 4; ++j) {
    int e = tid + j * 256;
    yr[e] = g[e] * (v[j] - mu) * rs + bta[e];
  }
}

// ---------------------------------------------------------------------------
extern "C" void kernel_launch(void* const* d_in, const int* in_sizes, int n_in,
                              void* d_out, int out_size, void* d_ws,
                              size_t ws_size, hipStream_t stream) {
  (void)in_sizes; (void)n_in; (void)out_size; (void)ws_size;
  const float* x      = (const float*)d_in[0];
  const float* Wq     = (const float*)d_in[1];
  const float* Wk     = (const float*)d_in[2];
  const float* Wv     = (const float*)d_in[3];
  const float* Wbeta  = (const float*)d_in[4];
  const float* b_beta = (const float*)d_in[5];
  const float* Wo     = (const float*)d_in[6];
  const float* b_o    = (const float*)d_in[7];
  const float* ln_g   = (const float*)d_in[8];
  const float* ln_b   = (const float*)d_in[9];
  float* out = (float*)d_out;

  char* ws = (char*)d_ws;
  size_t off = 0;
  auto alloc = [&](size_t bytes) {
    char* p = ws + off;
    off += (bytes + 255) & ~(size_t)255;
    return p;
  };
  unsigned short* xb   = (unsigned short*)alloc((size_t)MROWS * HID * 2);
  unsigned short* WkT  = (unsigned short*)alloc((size_t)HID * HID * 2);
  unsigned short* WoT  = (unsigned short*)alloc((size_t)HID * HID * 2);
  float* Wqs   = (float*)alloc((size_t)HID * NH * 4);
  float* Wvs   = (float*)alloc((size_t)HID * NH * 4);
  float* qs    = (float*)alloc((size_t)MROWS * NH * 4);
  float* vs    = (float*)alloc((size_t)MROWS * NH * 4);
  float* beta  = (float*)alloc((size_t)MROWS * NH * 4);
  unsigned short* kb = (unsigned short*)alloc((size_t)MROWS * HID * 2);
  unsigned short* ob = (unsigned short*)alloc((size_t)MROWS * HID * 2);
  float* Lend  = (float*)alloc((size_t)BQ * NH * NC * DH * 4);
  float* Aprod = (float*)alloc((size_t)BQ * NH * NC * 4);
  float* Hcs   = (float*)alloc((size_t)BQ * NH * NC * DH * 4);

  // 1. casts / transposes / head sums
  cast_x_kernel<<<(MROWS * HID / 4 + 255) / 256, 256, 0, stream>>>(
      x, xb, MROWS * HID / 4);
  wT_kernel<<<(HID * HID + 255) / 256, 256, 0, stream>>>(Wk, WkT, HID, HID);
  wT_kernel<<<(HID * HID + 255) / 256, 256, 0, stream>>>(Wo, WoT, HID, HID);
  headsum_kernel<<<(HID * NH + 255) / 256, 256, 0, stream>>>(Wq, Wqs);
  headsum_kernel<<<(HID * NH + 255) / 256, 256, 0, stream>>>(Wv, Wvs);

  // 2. fused small projection (qsum, vsum, beta)
  dim3 spB(64, 4);
  smallproj_kernel<<<MROWS / 4, spB, 0, stream>>>(x, Wqs, Wvs, Wbeta, b_beta,
                                                  qs, vs, beta);

  // 3. k = x @ Wk   (bf16 WMMA, bf16 output)
  dim3 gg(HID / BN, MROWS / BM);
  gemm_bf16_kernel<true><<<gg, 256, 0, stream>>>(xb, WkT, kb, nullptr, nullptr,
                                                 nullptr, MROWS, HID, HID);

  // 4. chunked linear scan -> o (bf16)
  scan_p1<<<BQ * NH * NC, DH, 0, stream>>>(kb, vs, beta, Lend, Aprod);
  scan_p2<<<BQ * NH, DH, 0, stream>>>(Lend, Aprod, Hcs);
  scan_p3<<<BQ * NH * NC, DH, 0, stream>>>(kb, qs, vs, beta, Hcs, ob);

  // 5. y = o @ Wo + b_o + x  (f32 into d_out)
  gemm_bf16_kernel<false><<<gg, 256, 0, stream>>>(ob, WoT, nullptr, out, b_o,
                                                  x, MROWS, HID, HID);

  // 6. in-place LayerNorm
  ln_kernel<<<MROWS, 256, 0, stream>>>(out, ln_g, ln_b);
}
